// nnet_82222853915411
// MI455X (gfx1250) — matrix operand order repaired and checked by reading, not compiled, not run on hardware
//
#include <hip/hip_runtime.h>
#include <hip/hip_bf16.h>

// ---------------------------------------------------------------------------
// Types for CDNA5 WMMA / TDM
// ---------------------------------------------------------------------------
typedef __attribute__((ext_vector_type(16))) __bf16 v16bf;
typedef __attribute__((ext_vector_type(8)))  float  v8f;
typedef __attribute__((ext_vector_type(4)))  unsigned int tdm_g0_t;
typedef __attribute__((ext_vector_type(8)))  int          tdm_g1_t;
typedef __attribute__((ext_vector_type(4)))  int          tdm_g23_t;
typedef __attribute__((ext_vector_type(8)))  int          tdm_g4_t;

#define IN_DIM 17
#define EMBED  512
#define HID    512
#define G4     2048   // 4 * HID

// ---------------------------------------------------------------------------
// bf16 helpers (round-to-nearest-even)
// ---------------------------------------------------------------------------
static __device__ __forceinline__ unsigned short f2bf(float f) {
  unsigned int u = __float_as_uint(f);
  u += 0x7FFFu + ((u >> 16) & 1u);
  return (unsigned short)(u >> 16);
}
static __device__ __forceinline__ float bf2f(unsigned short s) {
  return __uint_as_float(((unsigned int)s) << 16);
}
static __device__ __forceinline__ float bfl(unsigned int p) {
  return __uint_as_float(p << 16);
}
static __device__ __forceinline__ float bfh(unsigned int p) {
  return __uint_as_float(p & 0xFFFF0000u);
}

// ---------------------------------------------------------------------------
// Kernel 0: convert weights to bf16, zero h double-buffer + barrier counter.
// ---------------------------------------------------------------------------
__global__ void __launch_bounds__(256)
init_convert(const float* __restrict__ Wih, const float* __restrict__ Whh,
             unsigned short* __restrict__ Wih_bf, unsigned short* __restrict__ Whh_bf,
             float* __restrict__ hbuf, unsigned int* __restrict__ counter) {
  const unsigned int i = blockIdx.x * 256u + threadIdx.x;
  if (i < (unsigned int)(G4 * EMBED)) {
    Wih_bf[i] = f2bf(Wih[i]);
    Whh_bf[i] = f2bf(Whh[i]);
  }
  if (i < 2u * HID) hbuf[i] = 0.0f;   // h0 = 0 (both buffers)
  if (i == 0) *counter = 0u;
}

// ---------------------------------------------------------------------------
// Kernel 1: xg = relu(features @ W_emb + b_emb) @ W_ih^T + b_ih + b_hh
// 16 rows per block, 8 waves, each wave: 4 passes x 4 col-tiles. This shape
// is the one the backend pipelines cleanly: A fragments fully register-
// resident (k0 loop unrolled), B fragments double-buffered via clauses of
// global_load_b128, no scratch spills.
// ---------------------------------------------------------------------------
__global__ void __launch_bounds__(256)
embed_gates_wmma(const float* __restrict__ feats,
                 const float* __restrict__ W_emb,
                 const float* __restrict__ b_emb,
                 const unsigned short* __restrict__ Wih,   // [2048][512] bf16
                 const float* __restrict__ b_ih,
                 const float* __restrict__ b_hh,
                 unsigned short* __restrict__ xg) {        // [N][2048] bf16
  __shared__ float          fl[16 * 20];          // features tile (padded)
  __shared__ unsigned short xb[16 * EMBED];       // x tile, bf16 row-major

  const int tid = threadIdx.x;
  const int m0  = blockIdx.x << 4;

  // ---- stage features tile ----
  for (int p = tid; p < 16 * IN_DIM; p += 256) {
    const int r = p / IN_DIM, c = p - r * IN_DIM;
    fl[r * 20 + c] = feats[(size_t)(m0 + r) * IN_DIM + c];
  }
  __syncthreads();

  // ---- x = relu(f @ W_emb + b_emb), K0=17 on VALU, bf16 into LDS ----
  for (int p = tid; p < 16 * EMBED; p += 256) {
    const int r = p >> 9, e = p & 511;
    float acc = b_emb[e];
    #pragma unroll
    for (int i = 0; i < IN_DIM; ++i)
      acc = fmaf(fl[r * 20 + i], W_emb[i * EMBED + e], acc);
    xb[r * EMBED + e] = f2bf(fmaxf(acc, 0.0f));
  }
  __syncthreads();

  // ---- WMMA: xg tile = x_tile @ W_ih^T ----
  const int wave  = tid >> 5;
  const int lane  = tid & 31;
  const int lhalf = lane >> 4;   // K-half select (this code's contiguous k order)
  const int l15   = lane & 15;

  const v8f vzero = (v8f){0.f, 0.f, 0.f, 0.f, 0.f, 0.f, 0.f, 0.f};

  for (int pass = 0; pass < 4; ++pass) {
    v8f acc[4];
    #pragma unroll
    for (int c = 0; c < 4; ++c) acc[c] = vzero;

    const int nbase = pass * 512 + wave * 64;

    for (int k0 = 0; k0 < EMBED; k0 += 32) {
      // A fragment: lane<16 holds K {0..15}, lane>=16 holds {16..31}
      union { uint4 u[2]; v16bf v; } A;
      const int ka = k0 + (lhalf << 4);
      A.u[0] = *(const uint4*)&xb[l15 * EMBED + ka];
      A.u[1] = *(const uint4*)&xb[l15 * EMBED + ka + 8];

      #pragma unroll
      for (int c = 0; c < 4; ++c) {
        // B fragment: lane holds 16 contiguous K of W_ih row n (row-major => linear)
        const int n  = nbase + c * 16 + l15;
        const int kb = k0 + (lhalf << 4);
        union { uint4 u[2]; v16bf v; } B;
        const uint4* gp = (const uint4*)(Wih + (size_t)n * EMBED + kb);
        B.u[0] = gp[0];
        B.u[1] = gp[1];
        acc[c] = __builtin_amdgcn_wmma_f32_16x16x32_bf16(
                     false, A.v, false, B.v, (short)0, acc[c], false, false);
      }
    }

    // ---- epilogue: fold both biases, store bf16 (C/D layout: VGPR i ->
    //      M = i + 8*(lane>=16), N = lane&15) ----
    #pragma unroll
    for (int c = 0; c < 4; ++c) {
      const int col  = nbase + c * 16 + l15;
      const float bb = b_ih[col] + b_hh[col];
      #pragma unroll
      for (int i = 0; i < 8; ++i) {
        const int row = i + (lhalf << 3);
        xg[((size_t)(m0 + row) << 11) + col] = f2bf(acc[c][i] + bb);
      }
    }
  }
}

// ---------------------------------------------------------------------------
// Kernel 2: LSTM recurrence. 16 workgroups x 256 threads. WG w owns gate rows
// {g*512 + w*32 + j} of W_hh resident in LDS (128 KB bf16). The slice is
// brought in with 4 TDM tensor loads (each gate's 32 rows are contiguous:
// a flat 16384-element copy). h double-buffered in global; device-scope
// counter barrier per step.
// ---------------------------------------------------------------------------
__global__ void __launch_bounds__(256)
lstm_recurrence(const unsigned short* __restrict__ xg,   // [T][2048] bf16 (biases folded)
                const unsigned short* __restrict__ Whh,  // [2048][512] bf16
                float* __restrict__ hbuf,                // [2][512]
                unsigned int* __restrict__ counter,
                const int T) {
  extern __shared__ char smem[];
  unsigned short* Wl = (unsigned short*)smem;                       // [128][512] bf16
  float* hl = (float*)(smem + 128 * HID * sizeof(unsigned short));  // [512]
  float* gl = hl + HID;                                             // [128]

  const int tid = threadIdx.x;
  const int wg  = blockIdx.x;           // 0..15

  // ---- TDM preload of this WG's W_hh slice: 4 x 32KB contiguous tiles ----
  if (tid < 32) {                        // wave 0 issues the DMAs
    #pragma unroll
    for (int gate = 0; gate < 4; ++gate) {
      const unsigned long long ga =
          (unsigned long long)(const void*)(Whh + ((size_t)gate * HID + wg * 32) * HID);
      tdm_g0_t g0;
      g0[0] = 1u;                                        // count=1 (valid D#)
      g0[1] = (unsigned)(gate * 32 * HID * 2);           // lds_addr (bytes)
      g0[2] = (unsigned)(ga & 0xFFFFFFFFull);            // global_addr[31:0]
      g0[3] = (unsigned)((ga >> 32) & 0x01FFFFFFull)     // global_addr[56:32]
            | 0x80000000u;                               // type=2 ("image")
      tdm_g1_t g1;
      g1[0] = 0x10000;          // workgroup_mask=0, data_size=1 (2 bytes)
      g1[1] = 0x40000000;       // tensor_dim0 = 16384 (bits 79:48 -> [31:16])
      g1[2] = 0x00010000;       // tensor_dim1 = 1
      g1[3] = 0x40000000;       // tile_dim0 = 16384 (bits 127:112)
      g1[4] = 1;                // tile_dim1 = 1
      g1[5] = 16384;            // tensor_dim0_stride low 32
      g1[6] = 0;
      g1[7] = 0;
      const tdm_g23_t gz  = (tdm_g23_t)0;
      const tdm_g4_t  gz8 = (tdm_g4_t)0;
      __builtin_amdgcn_tensor_load_to_lds(g0, g1, gz, gz, gz8, 0);
    }
    __builtin_amdgcn_s_wait_tensorcnt(0);
  }
  float c = 0.0f;                        // c state: thread t<32 owns c[wg*32+t]
  __syncthreads();

  const int row   = tid >> 1;            // 0..127 (gate*32 + j)
  const int half  = tid & 1;
  const int kbase = half << 8;           // 0 or 256
  const int gcol  = (row >> 5) * HID + wg * 32 + (row & 31);

  for (int t = 0; t < T; ++t) {
    const int pb = t & 1;
    // ---- broadcast h into LDS (agent-scope loads bypass stale WGP cache) ----
    if (tid < 128) {
      #pragma unroll
      for (int q = 0; q < 4; ++q)
        hl[(tid << 2) + q] = __hip_atomic_load(&hbuf[pb * HID + (tid << 2) + q],
                                               __ATOMIC_RELAXED, __HIP_MEMORY_SCOPE_AGENT);
    }
    if (tid == 0 && t + 8 < T)
      __builtin_prefetch(xg + ((size_t)(t + 8) << 11), 0, 1);   // global_prefetch_b8
    __syncthreads();

    // ---- GEMV: 2 threads per gate row, 256 MACs each, bf16 weights from LDS ----
    const unsigned short* wrow = Wl + row * HID + kbase;
    const float* hp = hl + kbase;
    float sum = 0.0f;
    #pragma unroll 4
    for (int kk = 0; kk < 256; kk += 8) {
      const uint4  wv = *(const uint4*)(wrow + kk);
      const float4 h0 = *(const float4*)(hp + kk);
      const float4 h1 = *(const float4*)(hp + kk + 4);
      sum = fmaf(bfl(wv.x), h0.x, sum);  sum = fmaf(bfh(wv.x), h0.y, sum);
      sum = fmaf(bfl(wv.y), h0.z, sum);  sum = fmaf(bfh(wv.y), h0.w, sum);
      sum = fmaf(bfl(wv.z), h1.x, sum);  sum = fmaf(bfh(wv.z), h1.y, sum);
      sum = fmaf(bfl(wv.w), h1.z, sum);  sum = fmaf(bfh(wv.w), h1.w, sum);
    }
    sum += __shfl_xor(sum, 1);
    if (half == 0)
      gl[row] = sum + bf2f(xg[((size_t)t << 11) + gcol]);
    __syncthreads();

    // ---- gates + state update for this WG's 32 h elements ----
    if (tid < 32) {
      const float gi = gl[tid];
      const float gf = gl[32 + tid];
      const float gc = gl[64 + tid];
      const float go = gl[96 + tid];
      const float i_g = 1.0f / (1.0f + __expf(-gi));
      const float f_g = 1.0f / (1.0f + __expf(-gf));
      const float c_g = tanhf(gc);
      const float o_g = 1.0f / (1.0f + __expf(-go));
      c = fmaf(f_g, c, i_g * c_g);
      const float hn = o_g * tanhf(c);
      __hip_atomic_store(&hbuf[((t + 1) & 1) * HID + wg * 32 + tid], hn,
                         __ATOMIC_RELEASE, __HIP_MEMORY_SCOPE_AGENT);
    }
    __threadfence();
    __syncthreads();

    // ---- device-scope step barrier (monotonic counter, reset by init) ----
    if (tid == 0) {
      __hip_atomic_fetch_add(counter, 1u, __ATOMIC_ACQ_REL, __HIP_MEMORY_SCOPE_AGENT);
      const unsigned int target = 16u * (unsigned int)(t + 1);
      while (__hip_atomic_load(counter, __ATOMIC_ACQUIRE, __HIP_MEMORY_SCOPE_AGENT) < target)
        __builtin_amdgcn_s_sleep(1);
    }
    __syncthreads();
  }
}

// ---------------------------------------------------------------------------
// Kernel 3: head. One wave.
// ---------------------------------------------------------------------------
__global__ void __launch_bounds__(32)
head_kernel(const float* __restrict__ hfin,
            const float* __restrict__ w_mu, const float* __restrict__ b_mu,
            const float* __restrict__ w_ls, const float* __restrict__ b_ls,
            const float* __restrict__ eps, float* __restrict__ out) {
  const int lane = threadIdx.x;
  float mu = 0.0f, ls = 0.0f;
  for (int i = lane; i < HID; i += 32) {
    const float h = hfin[i];
    mu = fmaf(h, w_mu[i], mu);
    ls = fmaf(h, w_ls[i], ls);
  }
  #pragma unroll
  for (int s = 16; s > 0; s >>= 1) {
    mu += __shfl_xor(mu, s);
    ls += __shfl_xor(ls, s);
  }
  if (lane == 0) {
    const float sigma = expf(ls + b_ls[0]);
    const float k = mu + b_mu[0] + sigma * eps[0];
    out[0] = 0.5f * (tanhf(k) + 1.0f);
  }
}

// ---------------------------------------------------------------------------
// Launch
// ---------------------------------------------------------------------------
extern "C" void kernel_launch(void* const* d_in, const int* in_sizes, int n_in,
                              void* d_out, int out_size, void* d_ws, size_t ws_size,
                              hipStream_t stream) {
  (void)n_in; (void)out_size; (void)ws_size;
  const float* feats = (const float*)d_in[0];
  const float* eps   = (const float*)d_in[1];
  const float* W_emb = (const float*)d_in[2];
  const float* b_emb = (const float*)d_in[3];
  const float* W_ih  = (const float*)d_in[4];
  const float* W_hh  = (const float*)d_in[5];
  const float* b_ih  = (const float*)d_in[6];
  const float* b_hh  = (const float*)d_in[7];
  const float* w_mu  = (const float*)d_in[8];
  const float* b_mu  = (const float*)d_in[9];
  const float* w_ls  = (const float*)d_in[10];
  const float* b_ls  = (const float*)d_in[11];
  float* out = (float*)d_out;

  const int N = in_sizes[0] / IN_DIM;   // 65536 rows = LSTM sequence length

  // workspace layout
  char* ws = (char*)d_ws;
  unsigned short* Wih_bf = (unsigned short*)(ws);                           // 2 MB
  unsigned short* Whh_bf = (unsigned short*)(ws + (size_t)G4 * EMBED * 2);  // 2 MB
  unsigned short* xg     = (unsigned short*)(ws + (size_t)G4 * EMBED * 4);  // N*2048 bf16
  float*          hbuf   = (float*)(ws + (size_t)G4 * EMBED * 4 + (size_t)N * G4 * 2);
  unsigned int*   counter = (unsigned int*)((char*)hbuf + 2 * HID * sizeof(float));

  init_convert<<<(G4 * EMBED + 255) / 256, 256, 0, stream>>>(
      W_ih, W_hh, Wih_bf, Whh_bf, hbuf, counter);

  embed_gates_wmma<<<N / 16, 256, 0, stream>>>(
      feats, W_emb, b_emb, Wih_bf, b_ih, b_hh, xg);

  const size_t recLds = 128 * HID * sizeof(unsigned short)   // 128 KB W_hh slice
                      + HID * sizeof(float)                  // h broadcast
                      + 128 * sizeof(float);                 // gate staging
  (void)hipFuncSetAttribute((const void*)lstm_recurrence,
                            hipFuncAttributeMaxDynamicSharedMemorySize, (int)recLds);
  lstm_recurrence<<<16, 256, recLds, stream>>>(xg, Whh_bf, hbuf, counter, N);

  // N even => final h is in buffer 0
  head_kernel<<<1, 32, 0, stream>>>(hbuf, w_mu, b_mu, w_ls, b_ls, eps, out);
}
